// MultiHeadAttention_59261958751001
// MI455X (gfx1250) — compile-verified, matrix operand order checked by reading, not run on hardware
//
#include <hip/hip_runtime.h>
#include <hip/hip_bf16.h>
#include <stdint.h>

typedef __attribute__((ext_vector_type(16))) _Float16 v16h;
typedef __attribute__((ext_vector_type(8)))  _Float16 v8h;
typedef __attribute__((ext_vector_type(4)))  _Float16 v4h;
typedef __attribute__((ext_vector_type(8)))  float    v8f;
typedef __attribute__((ext_vector_type(4)))  float    v4f;
typedef __attribute__((ext_vector_type(4)))  int      v4i;

#define WMMA_F32_F16(a, b, c) \
    __builtin_amdgcn_wmma_f32_16x16x32_f16(false, (a), false, (b), (short)0, (c), false, false)

// CDNA5 async global->LDS path (ASYNCcnt), guarded so compile stays green on
// toolchains without the gfx1250 async builtins. Probe-confirmed: the builtin's
// pointer params are int4-typed; AS-qualified pointers implicitly decay to
// generic, so qualify explicitly (AS1 source, AS3 LDS dest).
#if defined(__AMDGCN__) && __has_builtin(__builtin_amdgcn_global_load_async_to_lds_b128) && \
    __has_builtin(__builtin_amdgcn_s_wait_asynccnt)
#define USE_ASYNC_LDS 1
#define AS1P(p) ((__attribute__((address_space(1))) v4i*)(p))
#define AS3P(p) ((__attribute__((address_space(3))) v4i*)(p))
#else
#define USE_ASYNC_LDS 0
#endif

__device__ __forceinline__ v16h mk16(v8h lo, v8h hi) {
    v16h r;
#pragma unroll
    for (int j = 0; j < 8; ++j) { r[j] = lo[j]; r[j + 8] = hi[j]; }
    return r;
}
__device__ __forceinline__ v8h ld8h(const _Float16* p) { return *(const v8h*)p; }
__device__ __forceinline__ v8f zero8() {
    v8f z;
#pragma unroll
    for (int j = 0; j < 8; ++j) z[j] = 0.0f;
    return z;
}

// ---------------------------------------------------------------- f32 -> f16 (vectorized, n % 4 == 0)
__global__ __launch_bounds__(256) void k_cvt_f16(const float* __restrict__ in,
                                                 _Float16* __restrict__ out, int n4) {
    int i = blockIdx.x * 256 + threadIdx.x;
    if (i < n4) {
        v4f f = ((const v4f*)in)[i];
        v4h h;
#pragma unroll
        for (int j = 0; j < 4; ++j) h[j] = (_Float16)f[j];
        ((v4h*)out)[i] = h;
    }
}

// ---------------------------------------------------------------- GEMM  C[M,N] = A[M,K] * B[N,K]^T + bias
// A row-major [M,K] f16, B row-major [N,K] f16 ("NT"), bias f32 per column.
// Block: 256 thr = 8 waves (4 along M x 2 along N). Wave tile 32x64 (2x4 wmma frags).
// Block tile: 128 (M) x 128 (N). K stepped by 32.
// QKV=true: epilogue scatters columns to per-head q (scaled 1/sqrt(64)) / k / v buffers
//           laid out [B*H, T, 64] f16. QKV=false: f32 [M,N] output into Cout.
template <bool QKV>
__global__ __launch_bounds__(256) void k_gemm_nt(const _Float16* __restrict__ A,
                                                 const _Float16* __restrict__ Bm,
                                                 const float* __restrict__ bias,
                                                 void* __restrict__ Cout,
                                                 _Float16* __restrict__ q,
                                                 _Float16* __restrict__ k,
                                                 _Float16* __restrict__ v,
                                                 int M, int N, int K) {
    const int lane = threadIdx.x & 31;
    const int wave = threadIdx.x >> 5;
    const int sel  = lane >> 4;   // which K-half group this lane holds
    const int l15  = lane & 15;

    const int mBase = blockIdx.y * 128 + (wave & 3) * 32;
    const int nBase = blockIdx.x * 128 + (wave >> 2) * 64;

    v8f acc[2][4];
#pragma unroll
    for (int i = 0; i < 2; ++i)
#pragma unroll
        for (int j = 0; j < 4; ++j) acc[i][j] = zero8();

    for (int k0 = 0; k0 < K; k0 += 32) {
        v16h af[2], bf[4];
#pragma unroll
        for (int mi = 0; mi < 2; ++mi) {
            const _Float16* p = A + (size_t)(mBase + mi * 16 + l15) * K + k0 + sel * 8;
            af[mi] = mk16(ld8h(p), ld8h(p + 16));
        }
#pragma unroll
        for (int ni = 0; ni < 4; ++ni) {
            const _Float16* p = Bm + (size_t)(nBase + ni * 16 + l15) * K + k0 + sel * 8;
            bf[ni] = mk16(ld8h(p), ld8h(p + 16));
        }
        if (k0 + 64 < K) {  // stream hint two K-steps ahead -> global_prefetch_b8
            __builtin_prefetch(A + (size_t)(mBase + l15) * K + k0 + 64, 0, 1);
            __builtin_prefetch(Bm + (size_t)(nBase + l15) * K + k0 + 64, 0, 1);
        }
#pragma unroll
        for (int mi = 0; mi < 2; ++mi)
#pragma unroll
            for (int ni = 0; ni < 4; ++ni)
                acc[mi][ni] = WMMA_F32_F16(af[mi], bf[ni], acc[mi][ni]);
    }

#pragma unroll
    for (int mi = 0; mi < 2; ++mi)
#pragma unroll
        for (int ni = 0; ni < 4; ++ni)
#pragma unroll
            for (int r = 0; r < 8; ++r) {
                const int row = mBase + mi * 16 + r + sel * 8;  // C layout: vgpr r -> M=r (+8 upper half)
                const int col = nBase + ni * 16 + l15;          // lane -> N
                const float val = acc[mi][ni][r] + bias[col];
                if (QKV) {
                    // row = b*2048 + t ; col = which*1024 + h*64 + d
                    const int b = row >> 11, t = row & 2047;
                    const int which = col >> 10, rem = col & 1023;
                    const int h = rem >> 6, d = rem & 63;
                    const size_t dst = (((size_t)(b * 16 + h) * 2048) + t) * 64 + d;
                    if (which == 0)      q[dst] = (_Float16)(val * 0.125f);  // 1/sqrt(64)
                    else if (which == 1) k[dst] = (_Float16)val;
                    else                 v[dst] = (_Float16)val;
                } else {
                    ((float*)Cout)[(size_t)row * N + col] = val;
                }
            }
}

// ---------------------------------------------------------------- flash attention
// grid.x = T/64 (query tiles), grid.y = B*H. 128 threads = 4 waves, 16 query rows each.
__global__ __launch_bounds__(128) void k_flash_attn(const _Float16* __restrict__ Q,
                                                    const _Float16* __restrict__ Kg,
                                                    const _Float16* __restrict__ Vg,
                                                    _Float16* __restrict__ Oh) {
    constexpr int T = 2048, DH = 64, LDS = 72;  // 72-half row stride (16B aligned, conflict pad)
    __shared__ _Float16 Ksh[64 * LDS];
    __shared__ _Float16 VTsh[64 * LDS];          // V transposed: [d][key]
    __shared__ _Float16 Psh[4 * 16 * LDS];       // per-wave P re-layout buffer

    const int lane = threadIdx.x & 31, wave = threadIdx.x >> 5;
    const int sel = lane >> 4, l15 = lane & 15;
    const int bh = blockIdx.y, b = bh >> 4, h = bh & 15;
    const int qBase = blockIdx.x * 64 + wave * 16;

    const _Float16* Qb = Q  + (size_t)bh * T * DH;
    const _Float16* Kb = Kg + (size_t)bh * T * DH;
    const _Float16* Vb = Vg + (size_t)bh * T * DH;

    // Q fragments (A-matrix, K-dim = head dim, two 16x32 frags cover Dh=64)
    v16h aq[2];
    {
        const int row = qBase + l15;
#pragma unroll
        for (int f = 0; f < 2; ++f) {
            const _Float16* p = Qb + (size_t)row * DH + f * 32 + sel * 8;
            aq[f] = mk16(ld8h(p), ld8h(p + 16));
        }
    }

    v8f o[4];
#pragma unroll
    for (int i = 0; i < 4; ++i) o[i] = zero8();
    float m[8], l[8];
#pragma unroll
    for (int r = 0; r < 8; ++r) { m[r] = -1e30f; l[r] = 0.0f; }

    _Float16* Pw = Psh + wave * 16 * LDS;

    for (int kt = 0; kt < T / 64; ++kt) {
        __syncthreads();
        // cooperative stage: K tile row-major (async direct-to-LDS), V tile transposed
        for (int c = threadIdx.x; c < 512; c += 128) {
            const int row = c >> 3, c8 = (c & 7) * 8;
            const size_t g = (size_t)(kt * 64 + row) * DH + c8;
#if USE_ASYNC_LDS
            __builtin_amdgcn_global_load_async_to_lds_b128(
                AS1P(Kb + g), AS3P(&Ksh[row * LDS + c8]), 0, 0);
#else
            *(v8h*)&Ksh[row * LDS + c8] = ld8h(Kb + g);
#endif
            v8h vv = ld8h(Vb + g);
#pragma unroll
            for (int j = 0; j < 8; ++j) VTsh[(c8 + j) * LDS + row] = vv[j];
        }
#if USE_ASYNC_LDS
        __builtin_amdgcn_s_wait_asynccnt(0);
#endif
        __syncthreads();

        // S = Q K^T : 16 x 64 tile as 4 fragments
        v8f s[4];
#pragma unroll
        for (int nf = 0; nf < 4; ++nf) {
            const _Float16* pk = &Ksh[(nf * 16 + l15) * LDS + sel * 8];
            v16h bk0 = mk16(ld8h(pk),      ld8h(pk + 16));   // d 0..31
            v16h bk1 = mk16(ld8h(pk + 32), ld8h(pk + 48));   // d 32..63
            v8f sv = zero8();
            sv = WMMA_F32_F16(aq[0], bk0, sv);
            sv = WMMA_F32_F16(aq[1], bk1, sv);
            s[nf] = sv;
        }

        // online softmax (rows live at m = r + sel*8; cols span 16 lanes -> shfl reduce)
        float p[4][8];
#pragma unroll
        for (int r = 0; r < 8; ++r) {
            float tmax = s[0][r];
#pragma unroll
            for (int nf = 1; nf < 4; ++nf) tmax = fmaxf(tmax, s[nf][r]);
#pragma unroll
            for (int off = 1; off < 16; off <<= 1)
                tmax = fmaxf(tmax, __shfl_xor(tmax, off, 32));
            const float mn = fmaxf(m[r], tmax);
            float rs = 0.0f;
#pragma unroll
            for (int nf = 0; nf < 4; ++nf) {
                float pv = __expf(s[nf][r] - mn);
                p[nf][r] = pv;
                rs += pv;
            }
#pragma unroll
            for (int off = 1; off < 16; off <<= 1)
                rs += __shfl_xor(rs, off, 32);
            const float sc = __expf(m[r] - mn);
            l[r] = l[r] * sc + rs;
            m[r] = mn;
#pragma unroll
            for (int ni = 0; ni < 4; ++ni) o[ni][r] *= sc;
        }

        // C-layout -> A-layout for P via this wave's private LDS slab (same-wave DS ops are in order)
#pragma unroll
        for (int nf = 0; nf < 4; ++nf)
#pragma unroll
            for (int r = 0; r < 8; ++r)
                Pw[(r + sel * 8) * LDS + nf * 16 + l15] = (_Float16)p[nf][r];

        v16h ap[2];
#pragma unroll
        for (int f = 0; f < 2; ++f) {
            const _Float16* pp = Pw + l15 * LDS + f * 32 + sel * 8;
            ap[f] = mk16(ld8h(pp), ld8h(pp + 16));
        }

        // O += P V  (B fragments read contiguously from transposed V tile)
#pragma unroll
        for (int ni = 0; ni < 4; ++ni) {
            const _Float16* pv = &VTsh[(ni * 16 + l15) * LDS + sel * 8];
            v16h bv0 = mk16(ld8h(pv),      ld8h(pv + 16));   // keys 0..31
            v16h bv1 = mk16(ld8h(pv + 32), ld8h(pv + 48));   // keys 32..63
            o[ni] = WMMA_F32_F16(ap[0], bv0, o[ni]);
            o[ni] = WMMA_F32_F16(ap[1], bv1, o[ni]);
        }
    }

    // normalize and write heads-concat layout [B*T, 1024] f16 for output projection
#pragma unroll
    for (int ni = 0; ni < 4; ++ni)
#pragma unroll
        for (int r = 0; r < 8; ++r) {
            const int t = qBase + r + sel * 8;
            const int col = h * 64 + ni * 16 + l15;
            Oh[((size_t)b * T + t) * 1024 + col] = (_Float16)(o[ni][r] / l[r]);
        }
}

// ---------------------------------------------------------------- launch
extern "C" void kernel_launch(void* const* d_in, const int* in_sizes, int n_in,
                              void* d_out, int out_size, void* d_ws, size_t ws_size,
                              hipStream_t stream) {
    (void)in_sizes; (void)n_in; (void)out_size; (void)ws_size;
    const float* x     = (const float*)d_in[0];   // [4,2048,1024]
    const float* W_qkv = (const float*)d_in[1];   // [3072,1024]
    const float* b_qkv = (const float*)d_in[2];   // [3072]
    const float* W_out = (const float*)d_in[3];   // [1024,1024]
    const float* b_out = (const float*)d_in[4];   // [1024]
    float* out = (float*)d_out;                   // [4,2048,1024]

    const int M = 8192, C = 1024, N3 = 3072;
    char* ws = (char*)d_ws;
    _Float16* xh  = (_Float16*)(ws);                        // 16 MB
    _Float16* wqh = (_Float16*)(ws + (size_t)16777216);     //  6 MB
    _Float16* woh = (_Float16*)(ws + (size_t)23068672);     //  2 MB
    _Float16* qh  = (_Float16*)(ws + (size_t)25165824);     // 16 MB
    _Float16* kh  = (_Float16*)(ws + (size_t)41943040);     // 16 MB
    _Float16* vh  = (_Float16*)(ws + (size_t)58720256);     // 16 MB
    _Float16* oh  = (_Float16*)(ws + (size_t)75497472);     // 16 MB

    k_cvt_f16<<<(M * C / 4 + 255) / 256, 256, 0, stream>>>(x, xh, M * C / 4);
    k_cvt_f16<<<(N3 * C / 4 + 255) / 256, 256, 0, stream>>>(W_qkv, wqh, N3 * C / 4);
    k_cvt_f16<<<(C * C / 4 + 255) / 256, 256, 0, stream>>>(W_out, woh, C * C / 4);

    // QKV projection with fused head-scatter epilogue
    k_gemm_nt<true><<<dim3(N3 / 128, M / 128), 256, 0, stream>>>(
        xh, wqh, b_qkv, nullptr, qh, kh, vh, M, N3, C);

    k_flash_attn<<<dim3(2048 / 64, 64), 128, 0, stream>>>(qh, kh, vh, oh);

    // Output projection, f32 result + bias straight into d_out
    k_gemm_nt<false><<<dim3(C / 128, M / 128), 256, 0, stream>>>(
        oh, woh, b_out, out, nullptr, nullptr, nullptr, M, C, C);
}
